// Decoder_33208687133135
// MI455X (gfx1250) — compile-verified, matrix operand order checked by reading, not run on hardware
//
#include <hip/hip_runtime.h>
#include <cstddef>

// ---------------------------------------------------------------------------
// Types for CDNA5 WMMA (wave32): v_wmma_f32_16x16x32_bf16
// ---------------------------------------------------------------------------
typedef unsigned short ushort_t;
typedef __attribute__((ext_vector_type(16))) __bf16 v16bf;
typedef __attribute__((ext_vector_type(8)))  float  v8f;

union FragAB {            // 16 bf16 per lane = 8 VGPRs, filled by two b128 loads
    v16bf v;
    uint4 q[2];
};

__device__ __forceinline__ ushort_t f32_to_bf16(float f) {
    union { float f; unsigned u; } v; v.f = f;
    unsigned r = v.u + 0x7FFFu + ((v.u >> 16) & 1u);   // round-to-nearest-even
    return (ushort_t)(r >> 16);
}
__device__ __forceinline__ float bf16_to_f32(ushort_t u) {
    union { unsigned u; float f; } v; v.u = ((unsigned)u) << 16;
    return v.f;
}

// Branch-free tanh using hardware transcendentals:
//   tanh(x) = 1 - 2/(exp(2x)+1)
// Clamp to +-12 (tanh(+-12) rounds to +-1 in f32) so exp can never overflow
// and no divergent branch is needed. Compiles to v_med3 + v_exp + v_rcp.
__device__ __forceinline__ float fast_tanh(float x) {
    float xc = __builtin_fminf(__builtin_fmaxf(x, -12.0f), 12.0f);
    float e  = __expf(2.0f * xc);                      // v_exp_f32 path
    return 1.0f - 2.0f * __builtin_amdgcn_rcpf(e + 1.0f);
}

// ---------------------------------------------------------------------------
// Prep kernels: weights -> transposed layouts (B-fragment friendly)
//   WT[n*K + k] = W[k*N + n]
// ---------------------------------------------------------------------------
__global__ void k_transpose_bf16(const float* __restrict__ W, ushort_t* __restrict__ WT,
                                 int K, int N) {
    int idx = blockIdx.x * blockDim.x + threadIdx.x;
    if (idx >= K * N) return;
    int n = idx / K;
    int k = idx - n * K;
    WT[idx] = f32_to_bf16(W[(size_t)k * N + n]);
}

__global__ void k_transpose_f32(const float* __restrict__ W, float* __restrict__ WT,
                                int K, int N) {
    int idx = blockIdx.x * blockDim.x + threadIdx.x;
    if (idx >= K * N) return;
    int n = idx / K;
    int k = idx - n * K;
    WT[idx] = W[(size_t)k * N + n];
}

// ---------------------------------------------------------------------------
// Fused MLP pipeline. Tile: 64 rows per workgroup, 256 threads = 8 waves.
// LDS = 2 x 64KB activation buffers + 512B  -> 2 workgroups per WGP (320KB).
// Each wave covers ALL 64 M-rows (4 WMMA tiles) x 64 N-cols (wave * 64).
// ---------------------------------------------------------------------------
#define ROWS_PER_WG 64
#define HDIM        512

// Stage Z (bf16) into LDS buffer (ROWS_PER_WG x 64, stride 64).
// Z[r, l] = koopman[b, l, d] with global row R = b*64 + d.
__device__ __forceinline__ void stage_z(const float* __restrict__ koop,
                                        ushort_t* buf, int wgRow, int tid) {
    for (int idx = tid; idx < ROWS_PER_WG * 64; idx += 256) {
        int r = idx >> 6;
        int l = idx & 63;
        int R = wgRow + r;
        int b = R >> 6;
        int d = R & 63;
        buf[idx] = f32_to_bf16(koop[(size_t)b * 4096 + (size_t)l * 64 + d]);
    }
}

// One dense layer: dst(64x512, LDS bf16) = tanh(src(64xK) @ W(KxN) + bias)
// WT is the transposed bf16 weight (N x K, row-major in K).
__device__ __forceinline__ void mlp_layer(const ushort_t* src, ushort_t* dst,
                                          const ushort_t* __restrict__ WT,
                                          const float* __restrict__ bias,
                                          int Kdim, int srcStride, int tid) {
    const int lane  = tid & 31;
    const int wave  = tid >> 5;
    const int lo16  = lane & 15;
    const int half  = lane >> 4;
    const int nBase = wave * 64;            // 8 waves x 64 cols = 512

    for (int nt = 0; nt < 4; ++nt) {
        const int nG   = nBase + nt * 16;
        const int colG = nG + lo16;         // this lane's output column (C layout N)
        const float bv = bias[colG];

        v8f acc0 = {0.f,0.f,0.f,0.f,0.f,0.f,0.f,0.f};
        v8f acc1 = acc0, acc2 = acc0, acc3 = acc0;

        const ushort_t* bptr = WT + (size_t)colG * Kdim + half * 8;

        for (int k0 = 0; k0 < Kdim; k0 += 32) {
            FragAB fb;      // B fragment: contiguous K for this lane's column
            fb.q[0] = *(const uint4*)(bptr + k0);
            fb.q[1] = *(const uint4*)(bptr + k0 + 16);

            const ushort_t* ap = src + (size_t)lo16 * srcStride + k0 + half * 8;

            FragAB fa;      // A fragment per ISA 16-bit A layout (two b128 loads)
            fa.q[0] = *(const uint4*)(ap);
            fa.q[1] = *(const uint4*)(ap + 16);
            acc0 = __builtin_amdgcn_wmma_f32_16x16x32_bf16(false, fa.v, false, fb.v,
                                                           (short)0, acc0, false, false);

            fa.q[0] = *(const uint4*)(ap + 16 * srcStride);
            fa.q[1] = *(const uint4*)(ap + 16 * srcStride + 16);
            acc1 = __builtin_amdgcn_wmma_f32_16x16x32_bf16(false, fa.v, false, fb.v,
                                                           (short)0, acc1, false, false);

            fa.q[0] = *(const uint4*)(ap + 32 * srcStride);
            fa.q[1] = *(const uint4*)(ap + 32 * srcStride + 16);
            acc2 = __builtin_amdgcn_wmma_f32_16x16x32_bf16(false, fa.v, false, fb.v,
                                                           (short)0, acc2, false, false);

            fa.q[0] = *(const uint4*)(ap + 48 * srcStride);
            fa.q[1] = *(const uint4*)(ap + 48 * srcStride + 16);
            acc3 = __builtin_amdgcn_wmma_f32_16x16x32_bf16(false, fa.v, false, fb.v,
                                                           (short)0, acc3, false, false);
        }

        // Epilogue: bias + branch-free tanh + bf16, scatter per C/D layout
        //   row = mt*16 + half*8 + i ; col = colG
        v8f accs[4] = {acc0, acc1, acc2, acc3};
#pragma unroll
        for (int mt = 0; mt < 4; ++mt) {
            const int rowBase = mt * 16 + half * 8;
#pragma unroll
            for (int i = 0; i < 8; ++i) {
                float v = fast_tanh(accs[mt][i] + bv);
                dst[(size_t)(rowBase + i) * HDIM + colG] = f32_to_bf16(v);
            }
        }
    }
}

// Diagonal final layer: out[r] = h3[r,:] . W4T[d,:] + b4[d], d = (wgRow+r)%64.
// One wave per 8 rows, lanes stride K, shuffle-reduce.
__device__ __forceinline__ void diag_stage(const ushort_t* buf,
                                           const float* __restrict__ W4T,
                                           const float* __restrict__ b4,
                                           float* outArr, int wgRow, int tid) {
    const int lane = tid & 31;
    const int wave = tid >> 5;
#pragma unroll 1
    for (int rr = 0; rr < ROWS_PER_WG / 8; ++rr) {
        const int r = wave * (ROWS_PER_WG / 8) + rr;
        const int d = (wgRow + r) & 63;
        const ushort_t* arow = buf + (size_t)r * HDIM;
        const float*    wrow = W4T + (size_t)d * HDIM;
        float acc = 0.f;
#pragma unroll
        for (int j = 0; j < 16; ++j) {
            int k = lane + j * 32;
            acc += bf16_to_f32(arow[k]) * wrow[k];
        }
#pragma unroll
        for (int off = 16; off > 0; off >>= 1)
            acc += __shfl_xor(acc, off, 32);
        if (lane == 0) outArr[r] = acc + b4[d];
    }
}

__global__ void __launch_bounds__(256)
k_fused_koopman(const float* __restrict__ x, const float* __restrict__ koop,
                const ushort_t* sW1T, const ushort_t* sW2T, const ushort_t* sW3T,
                const float* sW4T,
                const float* sb1, const float* sb2, const float* sb3, const float* sb4,
                const ushort_t* tW1T, const ushort_t* tW2T, const ushort_t* tW3T,
                const float* tW4T,
                const float* tb1, const float* tb2, const float* tb3, const float* tb4,
                float* __restrict__ out) {
    extern __shared__ char smem[];
    ushort_t* bufA = (ushort_t*)smem;                       // 64*512 bf16 = 64KB
    ushort_t* bufB = bufA + ROWS_PER_WG * HDIM;             // 64KB
    float*    dsArr = (float*)(bufB + ROWS_PER_WG * HDIM);  // 256B
    float*    dtArr = dsArr + ROWS_PER_WG;                  // 256B

    const int tid   = threadIdx.x;
    const int wgRow = blockIdx.x * ROWS_PER_WG;

    // ---- scale MLP (s) ----
    stage_z(koop, bufB, wgRow, tid);
    __syncthreads();
    mlp_layer(bufB, bufA, sW1T, sb1, 64, 64, tid);
    __syncthreads();
    mlp_layer(bufA, bufB, sW2T, sb2, 512, 512, tid);
    __syncthreads();
    mlp_layer(bufB, bufA, sW3T, sb3, 512, 512, tid);
    __syncthreads();
    diag_stage(bufA, sW4T, sb4, dsArr, wgRow, tid);   // reads bufA
    stage_z(koop, bufB, wgRow, tid);                  // writes bufB (disjoint)
    __syncthreads();

    // ---- translation MLP (t) ----
    mlp_layer(bufB, bufA, tW1T, tb1, 64, 64, tid);
    __syncthreads();
    mlp_layer(bufA, bufB, tW2T, tb2, 512, 512, tid);
    __syncthreads();
    mlp_layer(bufB, bufA, tW3T, tb3, 512, 512, tid);
    __syncthreads();
    diag_stage(bufA, tW4T, tb4, dtArr, wgRow, tid);
    __syncthreads();

    // ---- elementwise epilogue: out = (x - dt) * exp(-ds) ----
    if (tid < ROWS_PER_WG) {
        const int R = wgRow + tid;
        out[R] = (x[R] - dtArr[tid]) * __expf(-dsArr[tid]);
    }
}

// ---------------------------------------------------------------------------
// Host-side launch
// ---------------------------------------------------------------------------
extern "C" void kernel_launch(void* const* d_in, const int* in_sizes, int n_in,
                              void* d_out, int out_size, void* d_ws, size_t ws_size,
                              hipStream_t stream) {
    (void)in_sizes; (void)n_in; (void)out_size; (void)ws_size;

    const float* x    = (const float*)d_in[0];
    const float* koop = (const float*)d_in[1];
    const float* sW1 = (const float*)d_in[2];  const float* sb1 = (const float*)d_in[3];
    const float* sW2 = (const float*)d_in[4];  const float* sb2 = (const float*)d_in[5];
    const float* sW3 = (const float*)d_in[6];  const float* sb3 = (const float*)d_in[7];
    const float* sW4 = (const float*)d_in[8];  const float* sb4 = (const float*)d_in[9];
    const float* tW1 = (const float*)d_in[10]; const float* tb1 = (const float*)d_in[11];
    const float* tW2 = (const float*)d_in[12]; const float* tb2 = (const float*)d_in[13];
    const float* tW3 = (const float*)d_in[14]; const float* tb3 = (const float*)d_in[15];
    const float* tW4 = (const float*)d_in[16]; const float* tb4 = (const float*)d_in[17];
    float* out = (float*)d_out;

    // Workspace partition (converted weights only, ~2.5 MB)
    char* ws = (char*)d_ws;
    ushort_t* sW1T = (ushort_t*)ws; ws += (size_t)64  * 512 * 2;
    ushort_t* sW2T = (ushort_t*)ws; ws += (size_t)512 * 512 * 2;
    ushort_t* sW3T = (ushort_t*)ws; ws += (size_t)512 * 512 * 2;
    float*    sW4T = (float*)ws;    ws += (size_t)512 * 64  * 4;
    ushort_t* tW1T = (ushort_t*)ws; ws += (size_t)64  * 512 * 2;
    ushort_t* tW2T = (ushort_t*)ws; ws += (size_t)512 * 512 * 2;
    ushort_t* tW3T = (ushort_t*)ws; ws += (size_t)512 * 512 * 2;
    float*    tW4T = (float*)ws;    ws += (size_t)512 * 64  * 4;

    const int thr = 256;
    // W1: K=64, N=512 ; W2/W3: K=512, N=512 ; W4: K=512, N=64
    k_transpose_bf16<<<(64 * 512 + thr - 1) / thr, thr, 0, stream>>>(sW1, sW1T, 64, 512);
    k_transpose_bf16<<<(512 * 512 + thr - 1) / thr, thr, 0, stream>>>(sW2, sW2T, 512, 512);
    k_transpose_bf16<<<(512 * 512 + thr - 1) / thr, thr, 0, stream>>>(sW3, sW3T, 512, 512);
    k_transpose_f32 <<<(512 * 64 + thr - 1) / thr, thr, 0, stream>>>(sW4, sW4T, 512, 64);
    k_transpose_bf16<<<(64 * 512 + thr - 1) / thr, thr, 0, stream>>>(tW1, tW1T, 64, 512);
    k_transpose_bf16<<<(512 * 512 + thr - 1) / thr, thr, 0, stream>>>(tW2, tW2T, 512, 512);
    k_transpose_bf16<<<(512 * 512 + thr - 1) / thr, thr, 0, stream>>>(tW3, tW3T, 512, 512);
    k_transpose_f32 <<<(512 * 64 + thr - 1) / thr, thr, 0, stream>>>(tW4, tW4T, 512, 64);

    // 2 x 64KB activation buffers + 2 x 64 f32 diag arrays -> 2 WGs per WGP
    const int smemBytes = 2 * ROWS_PER_WG * HDIM * 2 + 2 * ROWS_PER_WG * 4;
    (void)hipFuncSetAttribute((const void*)k_fused_koopman,
                              hipFuncAttributeMaxDynamicSharedMemorySize, smemBytes);

    const int M = 2048 * 64;                   // B*D rows
    const int grid = M / ROWS_PER_WG;          // 2048 workgroups
    k_fused_koopman<<<grid, thr, smemBytes, stream>>>(
        x, koop,
        sW1T, sW2T, sW3T, sW4T, sb1, sb2, sb3, sb4,
        tW1T, tW2T, tW3T, tW4T, tb1, tb2, tb3, tb4,
        out);
}
